// SAGE_26225070309438
// MI455X (gfx1250) — matrix operand order repaired and checked by reading, compiled, not run on hardware
//
#include <hip/hip_runtime.h>
#include <hip/hip_bf16.h>

// ---------------------------------------------------------------------------
// Types for CDNA5 WMMA (wave32): 16x16x32 bf16 -> f32
// ---------------------------------------------------------------------------
typedef __bf16 bf16_t;
typedef __attribute__((ext_vector_type(16))) __bf16 v16bf;
typedef __attribute__((ext_vector_type(8)))  __bf16 v8bf;
typedef __attribute__((ext_vector_type(8)))  float  v8f;

#define SHUF16(lo, hi) __builtin_shufflevector((lo), (hi), 0,1,2,3,4,5,6,7,8,9,10,11,12,13,14,15)

// ---------------------------------------------------------------------------
// Degree: deg[dst] += 1.0f per edge
// ---------------------------------------------------------------------------
__global__ __launch_bounds__(256)
void degree_k(const long long* __restrict__ ei, float* __restrict__ deg, int E_) {
    int e = blockIdx.x * 256 + threadIdx.x;
    if (e < E_) {
        int dst = (int)ei[(size_t)E_ + e];
        atomicAdd(&deg[dst], 1.0f);
    }
}

// deg -> 1/max(deg,1) in place
__global__ __launch_bounds__(256)
void deginv_k(float* __restrict__ deg, int Nn) {
    int i = blockIdx.x * 256 + threadIdx.x;
    if (i < Nn) deg[i] = 1.0f / fmaxf(deg[i], 1.0f);
}

// f32 -> bf16 elementwise
__global__ __launch_bounds__(256)
void cvt_bf16_k(const float* __restrict__ src, bf16_t* __restrict__ dst, int total) {
    int i = blockIdx.x * 256 + threadIdx.x;
    if (i < total) dst[i] = (bf16_t)src[i];
}

// W [K_ x N_] f32 row-major -> Wt [N_ x K_] bf16 (transposed, row-major)
__global__ __launch_bounds__(256)
void tconv_k(const float* __restrict__ W, bf16_t* __restrict__ Wt, int K_, int N_) {
    int i = blockIdx.x * 256 + threadIdx.x;
    if (i < K_ * N_) {
        int k = i / N_, n = i - k * N_;
        Wt[(size_t)n * K_ + k] = (bf16_t)W[i];
    }
}

// mean = agg * deg_inv, converted to bf16.  cshift = log2(C)
__global__ __launch_bounds__(256)
void mean_cvt_k(const float* __restrict__ agg, const float* __restrict__ dinv,
                bf16_t* __restrict__ mb, int cshift, int total) {
    int i = blockIdx.x * 256 + threadIdx.x;
    if (i < total) {
        int n = i >> cshift;
        mb[i] = (bf16_t)(agg[i] * dinv[n]);
    }
}

// ---------------------------------------------------------------------------
// Scatter-add, one wave32 per edge.
// Layer 1: x is f32, C=128 -> float4 per lane (4 ch/lane)
// ---------------------------------------------------------------------------
__global__ __launch_bounds__(256)
void scatter_f32_k(const float* __restrict__ x, const long long* __restrict__ ei,
                   float* __restrict__ agg, int E_) {
    int lane = threadIdx.x & 31;
    int e    = blockIdx.x * 8 + (threadIdx.x >> 5);
    if (e >= E_) return;
    int src = (int)ei[e];
    int dst = (int)ei[(size_t)E_ + e];
    const float4 v = *(const float4*)(x + (size_t)src * 128 + lane * 4);
    float* p = agg + (size_t)dst * 128 + lane * 4;
    atomicAdd(p + 0, v.x);
    atomicAdd(p + 1, v.y);
    atomicAdd(p + 2, v.z);
    atomicAdd(p + 3, v.w);
}

// Layer 2: h is bf16, C=256 -> v8bf per lane (8 ch/lane), f32 accumulation
__global__ __launch_bounds__(256)
void scatter_bf16_k(const bf16_t* __restrict__ h, const long long* __restrict__ ei,
                    float* __restrict__ agg, int E_) {
    int lane = threadIdx.x & 31;
    int e    = blockIdx.x * 8 + (threadIdx.x >> 5);
    if (e >= E_) return;
    int src = (int)ei[e];
    int dst = (int)ei[(size_t)E_ + e];
    v8bf v = *(const v8bf*)(h + (size_t)src * 256 + lane * 8);
    float* p = agg + (size_t)dst * 256 + lane * 8;
#pragma unroll
    for (int i = 0; i < 8; ++i) atomicAdd(p + i, (float)v[i]);
}

// ---------------------------------------------------------------------------
// WMMA GEMM: D = A1 @ B1t^T + A2 @ B2t^T + bias  [per 16x16 tile, wave32]
// A1,A2: [M x K] bf16 row-major; B1t,B2t: [NCOL x K] bf16 (pre-transposed).
// Fragment layouts per CDNA5 ISA 7.12.2.
// Block = 128 threads = 4 waves; one 16-row tile per block; each wave
// covers NCOL/16/4 column tiles.
// ---------------------------------------------------------------------------
template <int K, int NCOL, bool RELU, bool OUT_BF16>
__global__ __launch_bounds__(128)
void sage_gemm_k(const bf16_t* __restrict__ A1, const bf16_t* __restrict__ A2,
                 const bf16_t* __restrict__ B1t, const bf16_t* __restrict__ B2t,
                 const float* __restrict__ bias, void* __restrict__ Dst, int M) {
    const int lane  = threadIdx.x & 31;
    const int wv    = threadIdx.x >> 5;      // 0..3
    const int m0    = blockIdx.x * 16;
    const int lhalf = lane >> 4;             // 0 or 1
    const int l15   = lane & 15;
    int mrow = m0 + l15;
    if (mrow > M - 1) mrow = M - 1;          // clamp loads; EXEC stays all-1 for WMMA
    const bf16_t* a1row = A1 + (size_t)mrow * K;
    const bf16_t* a2row = A2 + (size_t)mrow * K;

    constexpr int NT  = NCOL / 16;           // column tiles
    constexpr int TPW = NT / 4;              // tiles per wave

#pragma unroll
    for (int t = 0; t < TPW; ++t) {
        const int ntile = wv * TPW + t;
        const int n0    = ntile * 16;
        const int ncol  = n0 + l15;
        const bf16_t* b1row = B1t + (size_t)ncol * K;
        const bf16_t* b2row = B2t + (size_t)ncol * K;

        v8f acc = {};
#pragma unroll
        for (int k0 = 0; k0 < K; k0 += 32) {
            // A fragment: element i -> K = (i<8 ? i : i+8) + lhalf*8 + k0
            const int ka = k0 + lhalf * 8;
            v16bf a1 = SHUF16(*(const v8bf*)(a1row + ka),
                              *(const v8bf*)(a1row + ka + 16));
            v16bf a2 = SHUF16(*(const v8bf*)(a2row + ka),
                              *(const v8bf*)(a2row + ka + 16));
            // B fragment: element i -> K = (i<8 ? i : i+8) + lhalf*8 + k0
            const int kb = k0 + lhalf * 8;
            v16bf b1 = SHUF16(*(const v8bf*)(b1row + kb),
                              *(const v8bf*)(b1row + kb + 16));
            v16bf b2 = SHUF16(*(const v8bf*)(b2row + kb),
                              *(const v8bf*)(b2row + kb + 16));
            acc = __builtin_amdgcn_wmma_f32_16x16x32_bf16(
                      false, a1, false, b1, (short)0, acc, false, false);
            acc = __builtin_amdgcn_wmma_f32_16x16x32_bf16(
                      false, a2, false, b2, (short)0, acc, false, false);
        }

        const float bv = bias[ncol];
#pragma unroll
        for (int v = 0; v < 8; ++v) {
            float r = acc[v] + bv;
            if (RELU) r = fmaxf(r, 0.0f);
            const int m = m0 + v + lhalf * 8;   // C/D layout: VGPR v -> M=v+lhalf*8
            if (m < M) {
                if (OUT_BF16)
                    ((bf16_t*)Dst)[(size_t)m * NCOL + ncol] = (bf16_t)r;
                else
                    ((float*)Dst)[(size_t)m * NCOL + ncol] = r;
            }
        }
    }
}

// ---------------------------------------------------------------------------
// log_softmax over 64 channels, one wave32 per row, in place on d_out
// ---------------------------------------------------------------------------
__global__ __launch_bounds__(256)
void logsoftmax_k(float* __restrict__ o, int Nn) {
    int lane = threadIdx.x & 31;
    int row  = blockIdx.x * 8 + (threadIdx.x >> 5);
    if (row >= Nn) return;
    float* p = o + (size_t)row * 64 + lane * 2;
    float2 v = *(const float2*)p;
    float mx = fmaxf(v.x, v.y);
#pragma unroll
    for (int off = 16; off > 0; off >>= 1) mx = fmaxf(mx, __shfl_xor(mx, off, 32));
    float s = __expf(v.x - mx) + __expf(v.y - mx);
#pragma unroll
    for (int off = 16; off > 0; off >>= 1) s += __shfl_xor(s, off, 32);
    float l = mx + __logf(s);
    float2 r;
    r.x = v.x - l;
    r.y = v.y - l;
    *(float2*)p = r;
}

// ---------------------------------------------------------------------------
// Launch
// ---------------------------------------------------------------------------
extern "C" void kernel_launch(void* const* d_in, const int* in_sizes, int n_in,
                              void* d_out, int out_size, void* d_ws, size_t ws_size,
                              hipStream_t stream) {
    const float*     x   = (const float*)d_in[0];
    const long long* ei  = (const long long*)d_in[1];
    const float*     W1l = (const float*)d_in[2];
    const float*     W1r = (const float*)d_in[3];
    const float*     b1  = (const float*)d_in[4];
    const float*     W2l = (const float*)d_in[5];
    const float*     W2r = (const float*)d_in[6];
    const float*     b2  = (const float*)d_in[7];
    float*           out = (float*)d_out;

    const int Nn = in_sizes[0] / 128;   // 100000
    const int Ee = in_sizes[1] / 2;     // 1600000
    const int IN_C = 128, HID_C = 256, OUT_C = 64;

    // Workspace carve-out (256B aligned)
    char*  ws   = (char*)d_ws;
    size_t off  = 0;
    auto alloc = [&](size_t bytes) -> char* {
        char* p = ws + off;
        off += (bytes + 255) & ~(size_t)255;
        return p;
    };
    float*  deg   = (float*)alloc((size_t)Nn * 4);                 // becomes deg_inv
    bf16_t* xb    = (bf16_t*)alloc((size_t)Nn * IN_C * 2);
    float*  agg   = (float*)alloc((size_t)Nn * HID_C * 4);         // shared L1/L2
    bf16_t* meanb = (bf16_t*)alloc((size_t)Nn * HID_C * 2);        // shared L1/L2
    bf16_t* hb    = (bf16_t*)alloc((size_t)Nn * HID_C * 2);
    bf16_t* wt1l  = (bf16_t*)alloc((size_t)IN_C * HID_C * 2);
    bf16_t* wt1r  = (bf16_t*)alloc((size_t)IN_C * HID_C * 2);
    bf16_t* wt2l  = (bf16_t*)alloc((size_t)HID_C * OUT_C * 2);
    bf16_t* wt2r  = (bf16_t*)alloc((size_t)HID_C * OUT_C * 2);
    (void)ws_size; (void)n_in; (void)out_size;

    // ---- degree / deg_inv ----
    hipMemsetAsync(deg, 0, (size_t)Nn * 4, stream);
    hipMemsetAsync(agg, 0, (size_t)Nn * IN_C * 4, stream);
    degree_k<<<(Ee + 255) / 256, 256, 0, stream>>>(ei, deg, Ee);
    deginv_k<<<(Nn + 255) / 256, 256, 0, stream>>>(deg, Nn);

    // ---- precision prep: x -> bf16, weights -> bf16 transposed ----
    {
        int tot = Nn * IN_C;
        cvt_bf16_k<<<(tot + 255) / 256, 256, 0, stream>>>(x, xb, tot);
    }
    tconv_k<<<(IN_C * HID_C + 255) / 256, 256, 0, stream>>>(W1l, wt1l, IN_C, HID_C);
    tconv_k<<<(IN_C * HID_C + 255) / 256, 256, 0, stream>>>(W1r, wt1r, IN_C, HID_C);
    tconv_k<<<(HID_C * OUT_C + 255) / 256, 256, 0, stream>>>(W2l, wt2l, HID_C, OUT_C);
    tconv_k<<<(HID_C * OUT_C + 255) / 256, 256, 0, stream>>>(W2r, wt2r, HID_C, OUT_C);

    // ---- layer 1: scatter-mean + GEMM (WMMA bf16) + bias + ReLU ----
    scatter_f32_k<<<(Ee + 7) / 8, 256, 0, stream>>>(x, ei, agg, Ee);
    {
        int tot = Nn * IN_C;
        mean_cvt_k<<<(tot + 255) / 256, 256, 0, stream>>>(agg, deg, meanb, 7, tot);
    }
    sage_gemm_k<128, 256, true, true><<<(Nn + 15) / 16, 128, 0, stream>>>(
        meanb, xb, wt1l, wt1r, b1, (void*)hb, Nn);

    // ---- layer 2: scatter-mean + GEMM (WMMA bf16) + bias -> d_out ----
    hipMemsetAsync(agg, 0, (size_t)Nn * HID_C * 4, stream);
    scatter_bf16_k<<<(Ee + 7) / 8, 256, 0, stream>>>(hb, ei, agg, Ee);
    {
        int tot = Nn * HID_C;
        mean_cvt_k<<<(tot + 255) / 256, 256, 0, stream>>>(agg, deg, meanb, 8, tot);
    }
    sage_gemm_k<256, 64, false, false><<<(Nn + 15) / 16, 128, 0, stream>>>(
        meanb, hb, wt2l, wt2r, b2, (void*)out, Nn);

    // ---- log_softmax in place on d_out ----
    logsoftmax_k<<<(Nn + 7) / 8, 256, 0, stream>>>(out, Nn);
}
